// meta_gwnet_54322746359896
// MI455X (gfx1250) — matrix operand order repaired and checked by reading, not compiled, not run on hardware
//
#include <hip/hip_runtime.h>
#include <cstdint>

typedef _Float16 h16;
typedef __attribute__((ext_vector_type(16))) _Float16 v16h;
typedef __attribute__((ext_vector_type(8)))  _Float16 v8h;
typedef __attribute__((ext_vector_type(8)))  float    v8f;

#define WMMA_F16(a,b,c) __builtin_amdgcn_wmma_f32_16x16x32_f16(false,(a),false,(b),(short)0,(c),false,false)

static __device__ __forceinline__ v16h cat8(v8h lo, v8h hi){
  return __builtin_shufflevector(lo, hi, 0,1,2,3,4,5,6,7,8,9,10,11,12,13,14,15);
}
// A-operand (16x32 f16) per-lane load: row ptr -> k in [half*16,+16)
static __device__ __forceinline__ v16h loadA32(const h16* __restrict__ row, int half){
  v8h lo = *(const v8h*)(row + half*16);
  v8h hi = *(const v8h*)(row + half*16 + 8);
  return cat8(lo, hi);
}
static __device__ __forceinline__ uint32_t pkh(float a, float b){
  union { h16 h[2]; uint32_t u; } p; p.h[0] = (h16)a; p.h[1] = (h16)b; return p.u;
}

// ---------------------------------------------------------------------------
// Generic WMMA GEMM: C = act(A[M,K] x B[K,Nc] + bias)
// A: f16 row-major (lda). B: BMODE0 = packed WMMA-B blocks (block(kb,nb) = 256
// dwords: lane*8+v), BMODE1 = f16 row-major (ldb) with Kvalid row guard.
// CMODE: 0 = f16 row-major, 1 = f32 row-major, 2 = packed-B-layout f16 (M%32==0).
// BIAS: 0 none, 1 bias[n], 2 tap-permuted bias[(n%H)*2+n/H], H=Nc/2.
// One wave per 16x16 tile, 8 waves/block. z = batch index.
// ---------------------------------------------------------------------------
template<int BMODE,int CMODE,int BIAS,int ACT>
__global__ __launch_bounds__(256)
void gemm_ws(const h16* __restrict__ A, int lda, long aB,
             const void* __restrict__ Bv, long bB, int ldb, int Kvalid,
             void* __restrict__ Cv, int ldc, long cB, int Mstore,
             const float* __restrict__ bias,
             int nMT, int nNT, int nKC)
{
  int lane = threadIdx.x & 31;
  int tile = blockIdx.x * 8 + (threadIdx.x >> 5);
  if (tile >= nMT * nNT) return;           // wave-uniform
  int mt = tile / nNT, nt = tile % nNT;
  int z = blockIdx.z;
  int half = lane >> 4, l16 = lane & 15;
  const h16* Arow = A + (long)z * aB + (long)(mt * 16 + l16) * lda;
  v8f acc = {};
  for (int kc = 0; kc < nKC; ++kc) {
    v16h a = loadA32(Arow + kc * 32, half);
    v16h b;
    if (BMODE == 0) {
      const uint32_t* bp = (const uint32_t*)Bv + (long)z * bB +
                           ((long)(kc * nNT + nt) * 32 + lane) * 8;
      b = *(const v16h*)bp;
    } else {
      const h16* Bm = (const h16*)Bv + (long)z * bB + nt * 16 + l16;
      #pragma unroll
      for (int v = 0; v < 8; ++v) {
        int k0 = kc * 32 + half * 16 + v * 2;
        b[2*v]   = (k0     < Kvalid) ? Bm[(long)k0 * ldb]       : (h16)0.f;
        b[2*v+1] = (k0 + 1 < Kvalid) ? Bm[(long)(k0 + 1) * ldb] : (h16)0.f;
      }
    }
    acc = WMMA_F16(a, b, acc);
  }
  int n = nt * 16 + l16;
  float vals[8];
  #pragma unroll
  for (int j = 0; j < 8; ++j) {
    float vv = acc[j];
    if (BIAS == 1) vv += bias[n];
    else if (BIAS == 2) { int H = nNT * 8; vv += bias[(n % H) * 2 + (n / H)]; }
    if (ACT == 1) vv = fmaxf(vv, 0.f);
    vals[j] = vv;
  }
  if (CMODE == 2) {
    int kb = mt >> 1, hm = mt & 1;
    uint32_t* dst = (uint32_t*)Cv + (long)z * cB +
                    ((long)(kb * nNT + nt) * 32 + (l16 + 16 * hm)) * 8 + 4 * half;
    #pragma unroll
    for (int jj = 0; jj < 4; ++jj) dst[jj] = pkh(vals[2*jj], vals[2*jj+1]);
  } else {
    #pragma unroll
    for (int j = 0; j < 8; ++j) {
      int m = mt * 16 + 8 * half + j;
      if (m < Mstore) {
        long off = (long)z * cB + (long)m * ldc + n;
        if (CMODE == 0) ((h16*)Cv)[off] = (h16)vals[j];
        else            ((float*)Cv)[off] = vals[j];
      }
    }
  }
}

// Pack a small f32 matrix [K,Nc] into WMMA-B blocks (f16).
// mode 0: src[k*srcld+n]; mode 1: tap-permute cols (n%cout)*2+n/cout;
// mode 2: transposed source src[n*srcld+k].
__global__ void pack_b(const float* __restrict__ src, int srcld, int K, int Nc,
                       int nNB, int mode, uint32_t* __restrict__ dst)
{
  int idx = blockIdx.x * blockDim.x + threadIdx.x;
  int total = ((K + 31) / 32) * nNB * 256;
  if (idx >= total) return;
  int kb = idx / (nNB * 256);
  int rem = idx % (nNB * 256);
  int nb = rem / 256, lr = rem % 256;
  int lane = lr >> 3, v = lr & 7;
  int n = nb * 16 + (lane & 15);
  int k0 = kb * 32 + (lane >> 4) * 16 + v * 2;
  float f0 = 0.f, f1 = 0.f;
  if (n < Nc) {
    if (mode == 2) {
      if (k0     < K) f0 = src[(long)n * srcld + k0];
      if (k0 + 1 < K) f1 = src[(long)n * srcld + k0 + 1];
    } else {
      int col = n;
      if (mode == 1) { int cout = Nc >> 1; col = (n % cout) * 2 + (n / cout); }
      if (k0     < K) f0 = src[(long)k0 * srcld + col];
      if (k0 + 1 < K) f1 = src[(long)(k0 + 1) * srcld + col];
    }
  }
  dst[idx] = pkh(f0, f1);
}

// -------------------------- misc small kernels -----------------------------
__global__ void k_f2h(const float* __restrict__ s, h16* __restrict__ d, long n){
  long i = (long)blockIdx.x * blockDim.x + threadIdx.x;
  if (i < n) d[i] = (h16)s[i];
}
__global__ void k_zero_f32(float* __restrict__ d, long n){
  long i = (long)blockIdx.x * blockDim.x + threadIdx.x;
  if (i < n) d[i] = 0.f;
}
__global__ void k_relu_h(const float* __restrict__ s, h16* __restrict__ d, long n){
  long i = (long)blockIdx.x * blockDim.x + threadIdx.x;
  if (i < n) d[i] = (h16)fmaxf(s[i], 0.f);
}
// AT[si][w (336)][v (352)] = supports[si][v][w], zero padded (A-operand for diffusion)
__global__ void k_build_at(const float* __restrict__ S, h16* __restrict__ AT,
                           int N, int NP, int KP){
  int idx = blockIdx.x * blockDim.x + threadIdx.x;
  int total = 2 * NP * KP;
  if (idx >= total) return;
  int si = idx / (NP * KP), rem = idx % (NP * KP);
  int w = rem / KP, v = rem % KP;
  float val = (w < N && v < N) ? S[(long)si * N * N + (long)v * N + w] : 0.f;
  AT[idx] = (h16)val;
}
// start conv application: xc[b,t,n,o] (f16)
__global__ void k_start_app(const float* __restrict__ inp, const float* __restrict__ Wst,
                            const float* __restrict__ bst, h16* __restrict__ xc,
                            int N, int T, long total){
  long idx = (long)blockIdx.x * blockDim.x + threadIdx.x;
  if (idx >= total) return;
  int o = idx & 31;
  long r = idx >> 5;                 // (b*T + t)*N + n
  int n = (int)(r % N); long bt = r / N; int t = (int)(bt % T); int b = (int)(bt / T);
  long node = (long)b * N + n;
  long ib = (((long)b * N + n) * T + t) * 2;
  float v = Wst[(node*2+0)*32 + o] * inp[ib] + Wst[(node*2+1)*32 + o] * inp[ib+1]
          + bst[node*32 + o];
  xc[idx] = (h16)v;
}

// gated temporal conv: per node, 8 WMMAs -> tanh(f)*sigmoid(g)
__global__ __launch_bounds__(256)
void k_fg_app(const h16* __restrict__ xc, const uint32_t* __restrict__ Wf,
              const uint32_t* __restrict__ Wg, const float* __restrict__ biasL,
              h16* __restrict__ gout, int Tin, int L, int dil, int N, int BN)
{
  int lane = threadIdx.x & 31;
  int node = blockIdx.x * 8 + (threadIdx.x >> 5);
  if (node >= BN) return;
  int b = node / N, n = node % N;
  int half = lane >> 4, l16 = lane & 15;
  int tr = (l16 < L) ? l16 : (L - 1);
  const h16* base0 = xc + ((long)(b * Tin + tr)       * N + n) * 32;
  const h16* base1 = xc + ((long)(b * Tin + tr + dil) * N + n) * 32;
  v16h a0 = loadA32(base0, half);
  v16h a1 = loadA32(base1, half);
  v8f accf[2], accg[2];
  #pragma unroll
  for (int ob = 0; ob < 2; ++ob) {
    v8f c = {};
    v16h b0 = *(const v16h*)(Wf + ((long)(node*4 + 0 + ob) * 32 + lane) * 8);
    v16h b1 = *(const v16h*)(Wf + ((long)(node*4 + 2 + ob) * 32 + lane) * 8);
    c = WMMA_F16(a0, b0, c);
    c = WMMA_F16(a1, b1, c);
    accf[ob] = c;
    v8f d = {};
    v16h g0 = *(const v16h*)(Wg + ((long)(node*4 + 0 + ob) * 32 + lane) * 8);
    v16h g1 = *(const v16h*)(Wg + ((long)(node*4 + 2 + ob) * 32 + lane) * 8);
    d = WMMA_F16(a0, g0, d);
    d = WMMA_F16(a1, g1, d);
    accg[ob] = d;
  }
  #pragma unroll
  for (int ob = 0; ob < 2; ++ob) {
    #pragma unroll
    for (int j = 0; j < 8; ++j) {
      int t = j + 8 * half;
      if (t < L) {
        int o = l16 + 16 * ob;
        float f = tanhf(accf[ob][j] + biasL[(long)node*352 + o]);
        float g = accg[ob][j] + biasL[(long)node*352 + 32 + o];
        g = 1.f / (1.f + expf(-g));
        gout[((long)(b * L + t) * N + n) * 32 + o] = (h16)(f * g);
      }
    }
  }
}

// skip conv at last timestep only (see analysis), accumulated into skipacc
__global__ __launch_bounds__(256)
void k_skip_app(const h16* __restrict__ g, const uint32_t* __restrict__ Wsk,
                const float* __restrict__ biasL, float* __restrict__ skipacc,
                int L, int N)
{
  __shared__ float gl[32];
  int node = blockIdx.x;
  int b = node / N, n = node % N;
  int o = threadIdx.x;
  if (o < 32) gl[o] = (float)g[((long)(b * L + (L - 1)) * N + n) * 32 + o];
  __syncthreads();
  float s = biasL[(long)node*352 + 64 + o];
  const uint32_t* Wb = Wsk + ((long)node * 16 + (o >> 4)) * 256 + (o & 15) * 8;
  #pragma unroll
  for (int ih = 0; ih < 2; ++ih) {
    #pragma unroll
    for (int v = 0; v < 8; ++v) {
      union { uint32_t u; h16 h[2]; } q; q.u = Wb[ih * 128 + v];
      s += (float)q.h[0] * gl[ih*16 + 2*v] + (float)q.h[1] * gl[ih*16 + 2*v + 1];
    }
  }
  skipacc[(long)node * 256 + o] += s;
}

// GCN application: K=160 over 5 pieces, 10 WMMAs/node, + bias + residual
__global__ __launch_bounds__(256)
void k_gcn_app(const h16* __restrict__ g, const h16* __restrict__ x1a,
               const h16* __restrict__ x2a, const h16* __restrict__ x1b,
               const h16* __restrict__ x2b, const uint32_t* __restrict__ Wgcn,
               const float* __restrict__ biasL, const h16* __restrict__ xres,
               float* __restrict__ xpre, int Tin, int L, int N, int BN)
{
  int lane = threadIdx.x & 31;
  int node = blockIdx.x * 8 + (threadIdx.x >> 5);
  if (node >= BN) return;
  int b = node / N, n = node % N;
  int half = lane >> 4, l16 = lane & 15;
  int tr = (l16 < L) ? l16 : (L - 1);
  long rb = ((long)(b * L + tr) * N + n) * 32;
  const h16* pc[5] = { g + rb, x1a + rb, x2a + rb, x1b + rb, x2b + rb };
  v16h A[5];
  #pragma unroll
  for (int p = 0; p < 5; ++p) A[p] = loadA32(pc[p], half);
  int toff = Tin - L;
  #pragma unroll
  for (int ob = 0; ob < 2; ++ob) {
    v8f c = {};
    #pragma unroll
    for (int kc = 0; kc < 5; ++kc) {
      v16h bb = *(const v16h*)(Wgcn + (((long)(node*5 + kc) * 2 + ob) * 32 + lane) * 8);
      c = WMMA_F16(A[kc], bb, c);
    }
    #pragma unroll
    for (int j = 0; j < 8; ++j) {
      int t = j + 8 * half;
      if (t < L) {
        int o = l16 + 16 * ob;
        float v = c[j] + biasL[(long)node*352 + 320 + o]
                + (float)xres[((long)(b * Tin + t + toff) * N + n) * 32 + o];
        xpre[((long)(b * L + t) * N + n) * 32 + o] = v;
      }
    }
  }
}

__global__ void k_bn_reduce(const float* __restrict__ x, long n, float* __restrict__ st){
  __shared__ float ss[64];
  if (threadIdx.x < 64) ss[threadIdx.x] = 0.f;
  __syncthreads();
  int c = threadIdx.x & 31;
  float s = 0.f, s2 = 0.f;
  for (long i = (long)blockIdx.x * blockDim.x + threadIdx.x; i < n;
       i += (long)gridDim.x * blockDim.x) { float v = x[i]; s += v; s2 += v*v; }
  atomicAdd(&ss[c], s); atomicAdd(&ss[32 + c], s2);
  __syncthreads();
  if (threadIdx.x < 64) atomicAdd(&st[threadIdx.x], ss[threadIdx.x]);
}
__global__ void k_bn_norm(const float* __restrict__ x, const float* __restrict__ st,
                          const float* __restrict__ gam, const float* __restrict__ bet,
                          h16* __restrict__ out, long n){
  long i = (long)blockIdx.x * blockDim.x + threadIdx.x;
  if (i >= n) return;
  int c = i & 31;
  float cnt = (float)(n >> 5);
  float m = st[c] / cnt;
  float var = st[32 + c] / cnt - m * m;
  out[i] = (h16)((x[i] - m) * rsqrtf(var + 1e-5f) * gam[c] + bet[c]);
}
__global__ void k_end2(const h16* __restrict__ e1, const float* __restrict__ w,
                       const float* __restrict__ bb, float* __restrict__ out, int BN){
  int idx = blockIdx.x * blockDim.x + threadIdx.x;
  if (idx >= BN * 6) return;
  int node = idx / 6, o = idx % 6;
  float s = bb[o];
  const h16* row = e1 + (long)node * 512;
  for (int k = 0; k < 512; ++k) s += (float)row[k] * w[o * 512 + k];
  out[idx] = s;
}

// ---------------------------------------------------------------------------
extern "C" void kernel_launch(void* const* d_in, const int* in_sizes, int n_in,
                              void* d_out, int out_size, void* d_ws, size_t ws_size,
                              hipStream_t stream)
{
  (void)in_sizes; (void)n_in; (void)out_size; (void)ws_size;
  const int B = 32, N = 325, BN = B * N, T0 = 13;
  const float* MK  = (const float*)d_in[0];
  const float* INP = (const float*)d_in[1];
  const float* SUP = (const float*)d_in[2];
  struct MC { const float *w1,*b1,*w2,*b2,*wb,*bb; };
  auto getmc = [&](int base){ return MC{ (const float*)d_in[base], (const float*)d_in[base+1],
                                         (const float*)d_in[base+2], (const float*)d_in[base+3],
                                         (const float*)d_in[base+4], (const float*)d_in[base+5] }; };
  const int STA = 3, FIL = 9, GAT = 57, SKP = 105, GCN = 153, BNB = 201, E1 = 217, E2 = 219;
  MC st = getmc(STA);
  const float* E1W = (const float*)d_in[E1];     const float* E1B = (const float*)d_in[E1+1];
  const float* E2W = (const float*)d_in[E2];     const float* E2B = (const float*)d_in[E2+1];

  // -------- workspace arena --------
  char* p = (char*)d_ws;
  auto alloc = [&](size_t bytes)->char* { char* r = p; p += (bytes + 255) & ~(size_t)255; return r; };
  h16*      mkh     = (h16*)     alloc((size_t)BN*32*2);
  h16*      AT      = (h16*)     alloc((size_t)2*336*352*2);
  uint32_t* w1p     = (uint32_t*)alloc((size_t)1*320*256*4);       // K=32, Nc<=5120
  uint32_t* w2p     = (uint32_t*)alloc((size_t)1*16*256*4);        // Nc<=256
  uint32_t* wbp     = (uint32_t*)alloc((size_t)1*16*256*4);
  uint32_t* e1p     = (uint32_t*)alloc((size_t)8*32*256*4);        // K=256, Nc=512
  h16*      H1      = (h16*)     alloc((size_t)BN*5120*2);
  uint32_t* Wfil    = (uint32_t*)alloc((size_t)BN*4*256*4);
  uint32_t* Wgate   = (uint32_t*)alloc((size_t)BN*4*256*4);
  uint32_t* Wskip   = (uint32_t*)alloc((size_t)BN*16*256*4);
  uint32_t* Wgcn    = (uint32_t*)alloc((size_t)BN*5*2*256*4);
  float*    biasL   = (float*)   alloc((size_t)BN*352*4);
  float*    startW  = (float*)   alloc((size_t)BN*2*32*4);
  float*    startB  = (float*)   alloc((size_t)BN*32*4);
  h16*      xcurA   = (h16*)     alloc((size_t)B*13*N*32*2);
  h16*      xcurB   = (h16*)     alloc((size_t)B*13*N*32*2);
  h16*      gbuf    = (h16*)     alloc((size_t)B*12*N*32*2);
  h16*      x1a     = (h16*)     alloc((size_t)B*12*N*32*2);
  h16*      x2a     = (h16*)     alloc((size_t)B*12*N*32*2);
  h16*      x1b     = (h16*)     alloc((size_t)B*12*N*32*2);
  h16*      x2b     = (h16*)     alloc((size_t)B*12*N*32*2);
  float*    xpre    = (float*)   alloc((size_t)B*12*N*32*4);
  float*    skipacc = (float*)   alloc((size_t)BN*256*4);
  h16*      skiph   = (h16*)     alloc((size_t)BN*256*2);
  h16*      e1out   = (h16*)     alloc((size_t)BN*512*2);
  float*    bnst    = (float*)   alloc(64*4);

  auto cdiv = [](long a, long b){ return (int)((a + b - 1) / b); };
  auto pack = [&](const float* src, int K, int Nc, int srcld, int mode, uint32_t* dst){
    int total = ((K + 31) / 32) * (Nc / 16) * 256;
    pack_b<<<cdiv(total,256), 256, 0, stream>>>(src, srcld, K, Nc, Nc/16, mode, dst);
  };
  auto tgrid = [&](int nMT, int nNT, int nz){ return dim3((unsigned)((nMT*nNT + 7)/8), 1, (unsigned)nz); };

  // -------- setup --------
  k_f2h<<<cdiv((long)BN*32,256),256,0,stream>>>(MK, mkh, (long)BN*32);
  k_build_at<<<cdiv(2L*336*352,256),256,0,stream>>>(SUP, AT, N, 336, 352);
  k_zero_f32<<<cdiv((long)BN*256,256),256,0,stream>>>(skipacc, (long)BN*256);

  // -------- start conv --------
  pack(st.w1, 32, 64, 64, 0, w1p);
  gemm_ws<0,0,1,0><<<tgrid(650,4,1),256,0,stream>>>(mkh,32,0, w1p,0,0,0, H1,64,0,BN, st.b1, 650,4,1);
  pack(st.w2, 32, 32, 32, 0, w2p);
  gemm_ws<0,1,1,0><<<tgrid(1300,2,1),256,0,stream>>>(H1,32,0, w2p,0,0,0, startW,32,0,2*BN, st.b2, 1300,2,1);
  pack(st.wb, 32, 32, 32, 0, wbp);
  gemm_ws<0,1,1,0><<<tgrid(650,2,1),256,0,stream>>>(mkh,32,0, wbp,0,0,0, startB,32,0,BN, st.bb, 650,2,1);
  {
    long tot = (long)B*T0*N*32;
    k_start_app<<<cdiv(tot,256),256,0,stream>>>(INP, startW, startB, xcurA, N, T0, tot);
  }

  // -------- layers --------
  const int dils[8] = {1,2,1,2,1,2,1,2};
  h16 *xc = xcurA, *xn = xcurB;
  int Tin = T0;
  for (int li = 0; li < 8; ++li) {
    int dil = dils[li], L = Tin - dil;
    MC f  = getmc(FIL + 6*li), ga = getmc(GAT + 6*li);
    MC sk = getmc(SKP + 6*li), gc = getmc(GCN + 6*li);

    // filter weights
    pack(f.w1, 32, 1024, 1024, 0, w1p);
    gemm_ws<0,0,1,0><<<tgrid(650,64,1),256,0,stream>>>(mkh,32,0, w1p,0,0,0, H1,1024,0,BN, f.b1, 650,64,1);
    pack(f.w2, 32, 64, 64, 1, w2p);
    gemm_ws<0,2,2,0><<<tgrid(20800,4,1),256,0,stream>>>(H1,32,0, w2p,0,0,0, Wfil,0,0,0, f.b2, 20800,4,1);
    // gate weights
    pack(ga.w1, 32, 1024, 1024, 0, w1p);
    gemm_ws<0,0,1,0><<<tgrid(650,64,1),256,0,stream>>>(mkh,32,0, w1p,0,0,0, H1,1024,0,BN, ga.b1, 650,64,1);
    pack(ga.w2, 32, 64, 64, 1, w2p);
    gemm_ws<0,2,2,0><<<tgrid(20800,4,1),256,0,stream>>>(H1,32,0, w2p,0,0,0, Wgate,0,0,0, ga.b2, 20800,4,1);
    // skip weights
    pack(sk.w1, 32, 1024, 1024, 0, w1p);
    gemm_ws<0,0,1,0><<<tgrid(650,64,1),256,0,stream>>>(mkh,32,0, w1p,0,0,0, H1,1024,0,BN, sk.b1, 650,64,1);
    pack(sk.w2, 32, 256, 256, 0, w2p);
    gemm_ws<0,2,1,0><<<tgrid(20800,16,1),256,0,stream>>>(H1,32,0, w2p,0,0,0, Wskip,0,0,0, sk.b2, 20800,16,1);
    // gcn weights
    pack(gc.w1, 32, 5120, 5120, 0, w1p);
    gemm_ws<0,0,1,0><<<tgrid(650,320,1),256,0,stream>>>(mkh,32,0, w1p,0,0,0, H1,5120,0,BN, gc.b1, 650,320,1);
    pack(gc.w2, 32, 32, 32, 0, w2p);
    gemm_ws<0,2,1,0><<<tgrid(104000,2,1),256,0,stream>>>(H1,32,0, w2p,0,0,0, Wgcn,0,0,0, gc.b2, 104000,2,1);
    // per-node biases -> biasL cols [f:0, g:32, skip:64, gcn:320]
    pack(f.wb, 32, 32, 32, 0, wbp);
    gemm_ws<0,1,1,0><<<tgrid(650,2,1),256,0,stream>>>(mkh,32,0, wbp,0,0,0, biasL+0,352,0,BN, f.bb, 650,2,1);
    pack(ga.wb, 32, 32, 32, 0, wbp);
    gemm_ws<0,1,1,0><<<tgrid(650,2,1),256,0,stream>>>(mkh,32,0, wbp,0,0,0, biasL+32,352,0,BN, ga.bb, 650,2,1);
    pack(sk.wb, 32, 256, 256, 0, wbp);
    gemm_ws<0,1,1,0><<<tgrid(650,16,1),256,0,stream>>>(mkh,32,0, wbp,0,0,0, biasL+64,352,0,BN, sk.bb, 650,16,1);
    pack(gc.wb, 32, 32, 32, 0, wbp);
    gemm_ws<0,1,1,0><<<tgrid(650,2,1),256,0,stream>>>(mkh,32,0, wbp,0,0,0, biasL+320,352,0,BN, gc.bb, 650,2,1);

    // gated temporal conv
    k_fg_app<<<BN/8,256,0,stream>>>(xc, Wfil, Wgate, biasL, gbuf, Tin, L, dil, N, BN);
    // skip (last timestep only)
    k_skip_app<<<BN,256,0,stream>>>(gbuf, Wskip, biasL, skipacc, L, N);
    // diffusion: per (b,t) GEMM X1 = A^T * G  (M=336, K=352(325 valid), Nc=32)
    int nz = B * L;
    long bstr = (long)N * 32;
    gemm_ws<1,0,0,0><<<tgrid(21,2,nz),256,0,stream>>>(AT,            352,0, gbuf,bstr,32,N, x1a,32,bstr,N, nullptr, 21,2,11);
    gemm_ws<1,0,0,0><<<tgrid(21,2,nz),256,0,stream>>>(AT,            352,0, x1a, bstr,32,N, x2a,32,bstr,N, nullptr, 21,2,11);
    gemm_ws<1,0,0,0><<<tgrid(21,2,nz),256,0,stream>>>(AT + 336*352,  352,0, gbuf,bstr,32,N, x1b,32,bstr,N, nullptr, 21,2,11);
    gemm_ws<1,0,0,0><<<tgrid(21,2,nz),256,0,stream>>>(AT + 336*352,  352,0, x1b, bstr,32,N, x2b,32,bstr,N, nullptr, 21,2,11);
    // gcn application + residual
    k_gcn_app<<<BN/8,256,0,stream>>>(gbuf,x1a,x2a,x1b,x2b, Wgcn, biasL, xc, xpre, Tin, L, N, BN);
    // batchnorm
    long cnt = (long)B*L*N*32;
    k_zero_f32<<<1,64,0,stream>>>(bnst, 64);
    k_bn_reduce<<<256,256,0,stream>>>(xpre, cnt, bnst);
    k_bn_norm<<<cdiv(cnt,256),256,0,stream>>>(xpre, bnst,
        (const float*)d_in[BNB + 2*li], (const float*)d_in[BNB + 2*li + 1], xn, cnt);
    h16* tmp = xc; xc = xn; xn = tmp;
    Tin = L;
  }

  // -------- head --------
  k_relu_h<<<cdiv((long)BN*256,256),256,0,stream>>>(skipacc, skiph, (long)BN*256);
  pack(E1W, 256, 512, 256, 2, e1p);   // transposed: B[k,c] = w[c,k]
  gemm_ws<0,0,1,1><<<tgrid(650,32,1),256,0,stream>>>(skiph,256,0, e1p,0,0,0, e1out,512,0,BN, E1B, 650,32,8);
  k_end2<<<cdiv((long)BN*6,256),256,0,stream>>>(e1out, E2W, E2B, (float*)d_out, BN);
}
